// CosineSimilarityAttentionModel_PreLN_OutputNorm_23373212024849
// MI455X (gfx1250) — matrix operand order repaired and checked by reading, not compiled, not run on hardware
//
#include <hip/hip_runtime.h>
#include <hip/hip_bf16.h>

// ---------------------------------------------------------------------------
// CosineSimilarityAttention + PreLN MLP, fused for MI455X (gfx1250, wave32)
// - bf16 WMMA (v_wmma_f32_16x16x32_bf16) for all GEMMs
// - flash-style attention: sim matrix never touches HBM
// - K/Q tiles staged with GLOBAL_LOAD_ASYNC_TO_LDS_B128 (ASYNCcnt) + prefetch
// ---------------------------------------------------------------------------

typedef __bf16 bf16;
typedef bf16  v16bf  __attribute__((ext_vector_type(16)));
typedef bf16  bf16x8 __attribute__((ext_vector_type(8)));
typedef float v8f    __attribute__((ext_vector_type(8)));

#define HDIM 512
#define BATCH 16
#define SEQ 2048
#define FDIM 1024
#define TQ 16
#define TK 32
#define NW 4
#define HW 128   /* H columns per wave in attention */

__device__ __forceinline__ v8f v8f_zero() {
    v8f z;
#pragma unroll
    for (int i = 0; i < 8; ++i) z[i] = 0.0f;
    return z;
}

__device__ __forceinline__ v8f wmma_bf16(v16bf a, v16bf b, v8f c) {
    // D = A(16x32 bf16) * B(32x16 bf16) + C(16x16 f32)
    return __builtin_amdgcn_wmma_f32_16x16x32_bf16(
        /*neg_a=*/false, a, /*neg_b=*/false, b,
        /*c_mod=*/(short)0, c, /*reuse_a=*/false, /*reuse_b=*/false);
}

// LDS byte offset (wave-relative) of a generic pointer into __shared__.
__device__ __forceinline__ unsigned lds_addr32(void* p) {
    return (unsigned)(size_t)(__attribute__((address_space(3))) void*)p;
}

// Async DMA: 16 bytes per lane, global -> LDS, tracked by ASYNCcnt.
// GV mode: 64-bit per-lane global address; VDST holds LDS byte offset.
__device__ __forceinline__ void async_cp16(unsigned lds_byte, const void* gsrc) {
    asm volatile("global_load_async_to_lds_b128 %0, %1, off"
                 :: "v"(lds_byte), "v"((unsigned long long)(size_t)gsrc)
                 : "memory");
}
__device__ __forceinline__ void async_wait0() {
    asm volatile("s_wait_asynccnt 0" ::: "memory");
}

// A-matrix fragment (16x32 bf16, row-major source, row stride arbitrary).
// Lane l: m = l&15, half = l>>4. Per ISA 7.12.2: VGPR0..3 hold K = half*8..+7,
// VGPR4..7 hold K = 16+half*8..+7  -> two contiguous 16B chunks at byte
// offsets half*16 and 32+half*16 from (row m, k0).
__device__ __forceinline__ v16bf load_afrag(const bf16* row_k0, int half) {
    union { v16bf v; bf16x8 h[2]; } u;
    const char* p = (const char*)row_k0;
    u.h[0] = *(const bf16x8*)(p + half * 16);
    u.h[1] = *(const bf16x8*)(p + 32 + half * 16);
    return u.v;
}

// B-matrix fragment (32x16 bf16) from a K-major operand: lane l (n = l&15,
// half = l>>4) reads K = half*8..+7 and K = 16+half*8..+7 from the row that
// stores column n contiguously in K (i.e., the transposed matrix).
__device__ __forceinline__ v16bf load_bfrag(const bf16* coln_k0, int half) {
    union { v16bf v; bf16x8 h[2]; } u;
    const bf16* p = coln_k0 + half * 8;
    u.h[0] = *(const bf16x8*)(p);
    u.h[1] = *(const bf16x8*)(p + 16);
    return u.v;
}

// ---------------------------------------------------------------------------
// Kernel 1: L2-normalize rows of text1/text2 -> bf16, plus raw bf16 V copy.
// ---------------------------------------------------------------------------
__global__ __launch_bounds__(256) void prep_qkv(const float* __restrict__ t1,
                                                const float* __restrict__ t2,
                                                bf16* __restrict__ qn,
                                                bf16* __restrict__ kn,
                                                bf16* __restrict__ vn) {
    __shared__ float red[256];
    const int RQ = BATCH * SEQ;
    int row = blockIdx.x;
    int t = threadIdx.x;
    bool isK = row >= RQ;
    const float* src = isK ? (t2 + (size_t)(row - RQ) * HDIM)
                           : (t1 + (size_t)row * HDIM);
    float v0 = src[t];
    float v1 = src[t + 256];
    red[t] = v0 * v0 + v1 * v1;
    __syncthreads();
    for (int s = 128; s > 0; s >>= 1) {
        if (t < s) red[t] += red[t + s];
        __syncthreads();
    }
    float inv = 1.0f / fmaxf(sqrtf(red[0]), 1e-12f);
    if (!isK) {
        bf16* d = qn + (size_t)row * HDIM;
        d[t] = (bf16)(v0 * inv);
        d[t + 256] = (bf16)(v1 * inv);
    } else {
        size_t off = (size_t)(row - RQ) * HDIM;
        kn[off + t] = (bf16)(v0 * inv);
        kn[off + t + 256] = (bf16)(v1 * inv);
        vn[off + t] = (bf16)v0;
        vn[off + t + 256] = (bf16)v1;
    }
}

// ---------------------------------------------------------------------------
// Kernel 2: transpose + bf16-convert W1 [512,1024] and W2 [1024,512].
// ---------------------------------------------------------------------------
__global__ __launch_bounds__(256) void wprep(const float* __restrict__ W1,
                                             const float* __restrict__ W2,
                                             bf16* __restrict__ W1t,
                                             bf16* __restrict__ W2t) {
    int idx = blockIdx.x * 256 + threadIdx.x;
    const int N1 = HDIM * FDIM;  // 524288
    if (idx < N1) {
        int k = idx / FDIM, n = idx % FDIM;
        W1t[(size_t)n * HDIM + k] = (bf16)W1[idx];
    } else {
        int i2 = idx - N1;
        int k = i2 / HDIM, n = i2 % HDIM;
        W2t[(size_t)n * FDIM + k] = (bf16)W2[i2];
    }
}

// ---------------------------------------------------------------------------
// Kernel 3: flash attention. 1 block = 4 waves = one (batch, 16-query) tile.
// ---------------------------------------------------------------------------
__global__ __launch_bounds__(128) void attn_kernel(const bf16* __restrict__ qn,
                                                   const bf16* __restrict__ kn,
                                                   const bf16* __restrict__ vn,
                                                   const int* __restrict__ mask2,
                                                   float* __restrict__ attn_out) {
    __shared__ bf16 lds_q[TQ * HDIM];        // 16 KB
    __shared__ bf16 lds_k[TK * HDIM];        // 32 KB (normalized keys, row-major)
    __shared__ bf16 lds_vt[HDIM * TK];       // 32 KB (V transposed: vt[h][k])
    __shared__ float lds_sim[NW][TQ * TK];   // 8 KB  (per-wave sim partials)
    __shared__ bf16 lds_p[NW][TQ * TK];      // 4 KB  (per-wave P tiles)

    const int b = blockIdx.x >> 7;     // / (SEQ/TQ)
    const int qt = blockIdx.x & 127;
    const int tid = threadIdx.x;
    const int wv = tid >> 5;
    const int lane = tid & 31;
    const int half = lane >> 4;
    const int l16 = lane & 15;

    // Stage Q tile (16x512 bf16 = 1024 x 16B chunks) via async DMA
    {
        const char* src = (const char*)(qn + ((size_t)b * SEQ + qt * TQ) * HDIM);
        unsigned dst = lds_addr32(lds_q);
        for (int i = tid; i < 1024; i += 128)
            async_cp16(dst + i * 16, src + i * 16);
    }

    float m_old[8], l_old[8];
#pragma unroll
    for (int j = 0; j < 8; ++j) { m_old[j] = -3.0e38f; l_old[j] = 0.0f; }
    v8f acc[8];
#pragma unroll
    for (int n = 0; n < 8; ++n) acc[n] = v8f_zero();

    const int* mrow = mask2 + b * SEQ;
    async_wait0();
    __syncthreads();

    for (int kt = 0; kt < SEQ / TK; ++kt) {
        const size_t base = ((size_t)b * SEQ + kt * TK) * HDIM;
        // ---- stage K rows via async DMA; V rows transposed manually ----
        {
            const char* ks = (const char*)(kn + base);
            unsigned kd = lds_addr32(lds_k);
            for (int i = tid; i < 2048; i += 128)
                async_cp16(kd + i * 16, ks + i * 16);

            const uint4* vs = (const uint4*)(vn + base);
            for (int i = tid; i < 2048; i += 128) {
                uint4 d = vs[i];
                int r = i >> 6;            // key row 0..31
                int h0 = (i & 63) << 3;    // h offset of this 8-elem chunk
                const bf16* e = (const bf16*)&d;
#pragma unroll
                for (int q = 0; q < 8; ++q) lds_vt[(h0 + q) * TK + r] = e[q];
            }
        }
        async_wait0();
        __syncthreads();

        // speculative prefetch of next K/V tiles into cache (32 KB each,
        // 128 threads x 256B stride)  -> global_prefetch_b8
        if (kt + 1 < SEQ / TK) {
            const char* nk = (const char*)(kn + base) + TK * HDIM * 2;
            const char* nv = (const char*)(vn + base) + TK * HDIM * 2;
            __builtin_prefetch(nk + tid * 256, 0, 0);
            __builtin_prefetch(nv + tid * 256, 0, 0);
        }

        // ---- partial sim over this wave's H chunk [wv*128, +128) ----
        v8f simA = v8f_zero(), simB = v8f_zero();
#pragma unroll
        for (int kc = 0; kc < 4; ++kc) {
            const int k0 = wv * HW + kc * 32;
            v16bf af = load_afrag(lds_q + l16 * HDIM + k0, half);
            v16bf bfr0 = load_bfrag(lds_k + (l16)*HDIM + k0, half);
            v16bf bfr1 = load_bfrag(lds_k + (16 + l16) * HDIM + k0, half);
            simA = wmma_bf16(af, bfr0, simA);
            simB = wmma_bf16(af, bfr1, simB);
        }
#pragma unroll
        for (int j = 0; j < 8; ++j) {
            const int r = j + half * 8;
            lds_sim[wv][r * TK + l16] = simA[j];
            lds_sim[wv][r * TK + 16 + l16] = simB[j];
        }
        __syncthreads();

        // ---- reduce partials across waves (redundant per wave) ----
        float sA[8], sB[8];
#pragma unroll
        for (int j = 0; j < 8; ++j) {
            const int r = j + half * 8;
            float a = 0.0f, c = 0.0f;
#pragma unroll
            for (int w = 0; w < NW; ++w) {
                a += lds_sim[w][r * TK + l16];
                c += lds_sim[w][r * TK + 16 + l16];
            }
            sA[j] = a; sB[j] = c;
        }
        const int mA = mrow[kt * TK + l16];
        const int mB = mrow[kt * TK + 16 + l16];
#pragma unroll
        for (int j = 0; j < 8; ++j) {
            if (mA == 0) sA[j] = -3.0e38f;
            if (mB == 0) sB[j] = -3.0e38f;
        }

        // ---- online softmax update ----
        float alpha[8];
#pragma unroll
        for (int j = 0; j < 8; ++j) {
            float mv = fmaxf(sA[j], sB[j]);
#pragma unroll
            for (int off = 1; off < 16; off <<= 1)
                mv = fmaxf(mv, __shfl_xor(mv, off, 32));
            const float mnew = fmaxf(m_old[j], mv);
            const float pa = __expf(sA[j] - mnew);
            const float pb = __expf(sB[j] - mnew);
            float rs = pa + pb;
#pragma unroll
            for (int off = 1; off < 16; off <<= 1)
                rs += __shfl_xor(rs, off, 32);
            alpha[j] = __expf(m_old[j] - mnew);
            l_old[j] = l_old[j] * alpha[j] + rs;
            m_old[j] = mnew;
            const int r = j + half * 8;
            lds_p[wv][r * TK + l16] = (bf16)pa;
            lds_p[wv][r * TK + 16 + l16] = (bf16)pb;
        }
#pragma unroll
        for (int n = 0; n < 8; ++n)
#pragma unroll
            for (int j = 0; j < 8; ++j) acc[n][j] *= alpha[j];

        // same-wave LDS round trip (P tile): enforce DS ordering
        asm volatile("s_wait_dscnt 0" ::: "memory");

        // ---- out += P(16x32) x V(32x128) ----
        v16bf pf = load_afrag(&lds_p[wv][l16 * TK], half);
#pragma unroll
        for (int n = 0; n < 8; ++n) {
            const int hc = wv * HW + n * 16;
            v16bf bfr = load_bfrag(lds_vt + (hc + l16) * TK, half);
            acc[n] = wmma_bf16(pf, bfr, acc[n]);
        }
        __syncthreads();
    }

    // ---- epilogue: divide by softmax denominator, write f32 ----
    float rl[8];
#pragma unroll
    for (int j = 0; j < 8; ++j) rl[j] = 1.0f / l_old[j];
    float* orow = attn_out + ((size_t)b * SEQ + qt * TQ) * HDIM;
#pragma unroll
    for (int n = 0; n < 8; ++n) {
        const int c = wv * HW + n * 16 + l16;
#pragma unroll
        for (int j = 0; j < 8; ++j) {
            const int r = j + half * 8;
            orow[(size_t)r * HDIM + c] = acc[n][j] * rl[j];
        }
    }
}

// ---------------------------------------------------------------------------
// Kernel 4/5: C[M,N] = act(A[M,K]bf16 * Bt[N,K]bf16^T + bias)
// block = 128 threads (4 waves); each wave does a 16x64 tile.
// ---------------------------------------------------------------------------
template <int RELU, int BF16OUT>
__global__ __launch_bounds__(128) void gemm_bf16(const bf16* __restrict__ A,
                                                 const bf16* __restrict__ Bt,
                                                 const float* __restrict__ bias,
                                                 float* __restrict__ Cf,
                                                 bf16* __restrict__ Cb,
                                                 int M, int N, int K) {
    const int wv = threadIdx.x >> 5;
    const int lane = threadIdx.x & 31;
    const int half = lane >> 4;
    const int l16 = lane & 15;
    const int m0 = blockIdx.x * 64 + wv * 16;
    const int n0 = blockIdx.y * 64;

    v8f acc[4];
#pragma unroll
    for (int n = 0; n < 4; ++n) acc[n] = v8f_zero();

#pragma unroll 4
    for (int k0 = 0; k0 < K; k0 += 32) {
        v16bf af = load_afrag(A + (size_t)(m0 + l16) * K + k0, half);
#pragma unroll
        for (int n = 0; n < 4; ++n) {
            v16bf bfr = load_bfrag(Bt + (size_t)(n0 + n * 16 + l16) * K + k0, half);
            acc[n] = wmma_bf16(af, bfr, acc[n]);
        }
    }
#pragma unroll
    for (int n = 0; n < 4; ++n) {
        const int c = n0 + n * 16 + l16;
        const float bv = bias[c];
#pragma unroll
        for (int j = 0; j < 8; ++j) {
            const int r = m0 + j + half * 8;
            float v = acc[n][j] + bv;
            if (RELU) v = fmaxf(v, 0.0f);
            if (BF16OUT) Cb[(size_t)r * N + c] = (bf16)v;
            else         Cf[(size_t)r * N + c] = v;
        }
    }
}

// ---------------------------------------------------------------------------
// Kernel 6/7: LayerNorm epilogues.
// MODE 0: na = LN(x); na_bf16 <- na; out <- na + text1   (resid)
// MODE 1: out += LN(x)                                   (final)
// ---------------------------------------------------------------------------
template <int MODE>
__global__ __launch_bounds__(256) void ln_kernel(const float* __restrict__ X,
                                                 const float* __restrict__ g,
                                                 const float* __restrict__ bta,
                                                 const float* __restrict__ resid_src,
                                                 float* __restrict__ out,
                                                 bf16* __restrict__ nab) {
    __shared__ float red[256];
    const int row = blockIdx.x;
    const int t = threadIdx.x;
    const float* x = X + (size_t)row * HDIM;
    float v0 = x[t], v1 = x[t + 256];
    red[t] = v0 + v1;
    __syncthreads();
    for (int s = 128; s > 0; s >>= 1) {
        if (t < s) red[t] += red[t + s];
        __syncthreads();
    }
    const float mu = red[0] * (1.0f / (float)HDIM);
    __syncthreads();
    const float d0 = v0 - mu, d1 = v1 - mu;
    red[t] = d0 * d0 + d1 * d1;
    __syncthreads();
    for (int s = 128; s > 0; s >>= 1) {
        if (t < s) red[t] += red[t + s];
        __syncthreads();
    }
    const float inv = rsqrtf(red[0] * (1.0f / (float)HDIM) + 1e-6f);
    const float y0 = d0 * inv * g[t] + bta[t];
    const float y1 = d1 * inv * g[t + 256] + bta[t + 256];
    const size_t off = (size_t)row * HDIM;
    if (MODE == 0) {
        nab[off + t] = (bf16)y0;
        nab[off + t + 256] = (bf16)y1;
        out[off + t] = y0 + resid_src[off + t];
        out[off + t + 256] = y1 + resid_src[off + t + 256];
    } else {
        out[off + t] += y0;
        out[off + t + 256] += y1;
    }
}

// ---------------------------------------------------------------------------
extern "C" void kernel_launch(void* const* d_in, const int* in_sizes, int n_in,
                              void* d_out, int out_size, void* d_ws, size_t ws_size,
                              hipStream_t stream) {
    (void)in_sizes; (void)n_in; (void)out_size; (void)ws_size;
    const float* t1   = (const float*)d_in[0];
    const float* t2   = (const float*)d_in[1];
    const int*   m2   = (const int*)d_in[3];   // text1_mask (d_in[2]) unused by ref
    const float* W1   = (const float*)d_in[4];
    const float* b1   = (const float*)d_in[5];
    const float* W2   = (const float*)d_in[6];
    const float* b2   = (const float*)d_in[7];
    const float* ln1g = (const float*)d_in[8];
    const float* ln1b = (const float*)d_in[9];
    const float* ln2g = (const float*)d_in[10];
    const float* ln2b = (const float*)d_in[11];
    float* out = (float*)d_out;

    const size_t S1H = (size_t)BATCH * SEQ * HDIM;  // 16.78M elems
    char* p = (char*)d_ws;
    bf16* qn  = (bf16*)p;  p += S1H * 2;
    bf16* kn  = (bf16*)p;  p += S1H * 2;
    bf16* vn  = (bf16*)p;  p += S1H * 2;
    bf16* w1t = (bf16*)p;  p += (size_t)HDIM * FDIM * 2;
    bf16* w2t = (bf16*)p;  p += (size_t)FDIM * HDIM * 2;
    float* attn_out = (float*)p; p += S1H * 4;      // reused as ff2 output
    bf16* nab = (bf16*)p; p += S1H * 2;
    bf16* h1  = (bf16*)p; p += (size_t)BATCH * SEQ * FDIM * 2;
    float* ff2 = attn_out;  // attn_out consumed by LN1 before FF2 writes

    const int M = BATCH * SEQ;  // 32768 rows

    prep_qkv<<<2 * M, 256, 0, stream>>>(t1, t2, qn, kn, vn);
    wprep<<<(2 * HDIM * FDIM) / 256, 256, 0, stream>>>(W1, W2, w1t, w2t);
    attn_kernel<<<BATCH * (SEQ / TQ), 128, 0, stream>>>(qn, kn, vn, m2, attn_out);
    ln_kernel<0><<<M, 256, 0, stream>>>(attn_out, ln1g, ln1b, t1, out, nab);
    gemm_bf16<1, 1><<<dim3(M / 64, FDIM / 64), 128, 0, stream>>>(
        nab, w1t, b1, nullptr, h1, M, FDIM, HDIM);
    gemm_bf16<0, 0><<<dim3(M / 64, HDIM / 64), 128, 0, stream>>>(
        h1, w2t, b2, ff2, nullptr, M, HDIM, FDIM);
    ln_kernel<1><<<M, 256, 0, stream>>>(ff2, ln2g, ln2b, nullptr, out, nullptr);
}
